// ParallelSelfAttention_87222195847658
// MI455X (gfx1250) — matrix operand order repaired and checked by reading, not compiled, not run on hardware
//
#include <hip/hip_runtime.h>
#include <hip/hip_bf16.h>

#define SEQ     2048
#define NBATCH  2
#define NHEADS  16
#define HDIM    128
#define SEGLEN  1024
#define QTILE   64      // q rows per workgroup
#define WTILE   16      // q rows per wave
#define KTILE   32      // k rows per block
#define SROW    (NBATCH*NHEADS*HDIM)   // 4096 floats: stride between s positions

typedef __attribute__((ext_vector_type(16))) __bf16       v16bf;
typedef __attribute__((ext_vector_type(4)))  __bf16       v4bf;
typedef __attribute__((ext_vector_type(8)))  float        v8f;
typedef __attribute__((ext_vector_type(4)))  float        v4f;
typedef __attribute__((ext_vector_type(4)))  unsigned int v4u;

// LDS layout (units: __bf16 elements)
#define KSTRIDE 136                      // 32 x 128 (+8 pad) -> 272B rows (16B aligned)
#define VSTRIDE 40                       // 128 x 32 (+8 pad) -> 80B rows  (16B aligned)
#define PSTRIDE 40                       // 16 x 32 (+8 pad)  -> 80B rows
#define LDS_K_OFF 0
#define LDS_V_OFF (32*KSTRIDE)
#define LDS_P_OFF (LDS_V_OFF + 128*VSTRIDE)
#define LDS_TOTAL (LDS_P_OFF + 4*16*PSTRIDE)

// DPP lane-permute of a float within 16-lane rows (all lanes active).
// GCNDPPCombine folds the mov_dpp into the consuming v_max/v_add.
#define DPPF(x, ctrl)                                                          \
    __builtin_bit_cast(float, __builtin_amdgcn_update_dpp(                     \
        0, __builtin_bit_cast(int, (x)), (ctrl), 0xf, 0xf, true))

// full 16-lane butterfly reduction (per DPP row): xor1, xor2, half_mirror, mirror
#define ROW_REDUCE_MAX(v)                                                      \
    do {                                                                       \
        (v) = fmaxf((v), DPPF((v), 0xB1));   /* quad_perm(1,0,3,2) */          \
        (v) = fmaxf((v), DPPF((v), 0x4E));   /* quad_perm(2,3,0,1) */          \
        (v) = fmaxf((v), DPPF((v), 0x141));  /* row_half_mirror    */          \
        (v) = fmaxf((v), DPPF((v), 0x140));  /* row_mirror         */          \
    } while (0)

#define ROW_REDUCE_SUM(v)                                                      \
    do {                                                                       \
        (v) += DPPF((v), 0xB1);                                                \
        (v) += DPPF((v), 0x4E);                                                \
        (v) += DPPF((v), 0x141);                                               \
        (v) += DPPF((v), 0x140);                                               \
    } while (0)

__global__ __launch_bounds__(128) void fa_fwd_kernel(
    const float* __restrict__ Q, const float* __restrict__ K,
    const float* __restrict__ V, const float* __restrict__ Bias,
    float* __restrict__ Out)
{
    __shared__ __align__(16) __bf16 lds[LDS_TOTAL];

    const int tid  = threadIdx.x;
    const int qblk = blockIdx.x;          // 0..31
    const int bh   = blockIdx.y;          // 0..31  (b*16 + h)
    const int b    = bh >> 4;
    const int h    = bh & 15;

    const int q_wg = qblk * QTILE;
    const int seg0 = q_wg & ~(SEGLEN - 1);
    const int nb   = (q_wg + QTILE - seg0) >> 5;   // #32-row k blocks

    const int lane = tid & 31;
    const int wave = tid >> 5;
    const int hlf  = lane >> 4;           // lane half (WMMA layout)
    const int ln   = lane & 15;
    const int q0   = q_wg + WTILE * wave; // this wave's first q row

    const long long bh_off = (long long)b * (NHEADS * HDIM) + (long long)h * HDIM;
    const float* gk_base = K + bh_off;
    const float* gv_base = V + bh_off;
    const float* bias_b  = Bias + (long long)b * SEQ * SEQ;

    __bf16* ldsK = lds + LDS_K_OFF;
    __bf16* ldsV = lds + LDS_V_OFF;
    __bf16* myP  = lds + LDS_P_OFF + wave * 16 * PSTRIDE;

    // ---------------- Q tile -> WMMA A layout, pre-scaled by 1/sqrt(D) ------
    const float scale = 0.08838834764831845f;
    v16bf Aq[4];
    {
        const float* gq = Q + (long long)(q0 + ln) * SROW + bh_off;
        #pragma unroll
        for (int c = 0; c < 4; ++c) {
            const float* p = gq + 32 * c + 8 * hlf;
            v4f f0 = *(const v4f*)(p);
            v4f f1 = *(const v4f*)(p + 4);
            v4f f2 = *(const v4f*)(p + 16);
            v4f f3 = *(const v4f*)(p + 20);
            #pragma unroll
            for (int e = 0; e < 4; ++e) {
                Aq[c][e]      = (__bf16)(f0[e] * scale);
                Aq[c][4 + e]  = (__bf16)(f1[e] * scale);
                Aq[c][8 + e]  = (__bf16)(f2[e] * scale);
                Aq[c][12 + e] = (__bf16)(f3[e] * scale);
            }
        }
    }

    // ---------------- accumulators & running softmax stats ------------------
    v8f acc[8];
    #pragma unroll
    for (int t = 0; t < 8; ++t)
        #pragma unroll
        for (int r = 0; r < 8; ++r) acc[t][r] = 0.0f;

    float mrow[8], srow[8];
    #pragma unroll
    for (int r = 0; r < 8; ++r) { mrow[r] = -__builtin_inff(); srow[r] = 0.0f; }

    const int krow = tid >> 2;   // 0..31 (K loader row)
    const int kq   = tid & 3;    // quarter of the 128-wide row

    for (int kb = 0; kb < nb; ++kb) {
        const int k0 = seg0 + 32 * kb;

        // ---- cooperative K tile load: 32 x 128 f32 -> bf16 row-major ------
        {
            const float* g = gk_base + (long long)(k0 + krow) * SROW + kq * 32;
            __bf16* d = ldsK + krow * KSTRIDE + kq * 32;
            #pragma unroll
            for (int i = 0; i < 8; ++i) {
                v4f f = *(const v4f*)(g + 4 * i);
                v4bf hh;
                hh[0] = (__bf16)f[0]; hh[1] = (__bf16)f[1];
                hh[2] = (__bf16)f[2]; hh[3] = (__bf16)f[3];
                *(v4bf*)(d + 4 * i) = hh;       // 8B store, aligned
            }
        }
        // ---- cooperative V tile load, transposed: VT[d][k] bf16 -----------
        #pragma unroll
        for (int i = 0; i < 4; ++i) {
            int c  = tid + 128 * i;             // 0..511
            int kp = c & 15;                    // k pair -> rows 2kp,2kp+1
            int dg = c >> 4;                    // 0..31  -> d cols 4dg..4dg+3
            const float* g = gv_base + (long long)(k0 + 2 * kp) * SROW + dg * 4;
            v4f f0 = *(const v4f*)(g);
            v4f f1 = *(const v4f*)(g + SROW);
            #pragma unroll
            for (int j = 0; j < 4; ++j) {
                union { __bf16 hh[2]; unsigned u; } pk;
                pk.hh[0] = (__bf16)f0[j];       // V[2kp  ][d]
                pk.hh[1] = (__bf16)f1[j];       // V[2kp+1][d]
                *(unsigned*)(ldsV + (dg * 4 + j) * VSTRIDE + 2 * kp) = pk.u;
            }
        }
        // prefetch next k-block's K/V rows into cache (gfx1250 global_prefetch)
        if (kb + 1 < nb) {
            __builtin_prefetch(gk_base + (long long)(k0 + 32 + krow) * SROW + kq * 32, 0, 3);
            __builtin_prefetch(gv_base + (long long)(k0 + 32 + krow) * SROW + kq * 32, 0, 3);
        }
        __syncthreads();

        if (k0 <= q0 + 15) {                    // any unmasked element for this wave?
            // ---- C initialized with the additive score bias ---------------
            v8f C0, C1;
            #pragma unroll
            for (int r = 0; r < 8; ++r) {
                const float* bp = bias_b + (long long)(q0 + r + 8 * hlf) * SEQ + k0 + ln;
                C0[r] = bp[0];
                C1[r] = bp[16];
            }
            // ---- S = (Q*scale) K^T + bias : 8 WMMAs over D=128 ------------
            #pragma unroll
            for (int c = 0; c < 4; ++c) {
                v16bf B0, B1;
                const __bf16* r0 = ldsK + ln * KSTRIDE + 32 * c + 8 * hlf;
                ((v4u*)&B0)[0] = *(const v4u*)(r0);
                ((v4u*)&B0)[1] = *(const v4u*)(r0 + 16);
                const __bf16* r1 = r0 + 16 * KSTRIDE;
                ((v4u*)&B1)[0] = *(const v4u*)(r1);
                ((v4u*)&B1)[1] = *(const v4u*)(r1 + 16);
                C0 = __builtin_amdgcn_wmma_f32_16x16x32_bf16(false, Aq[c], false, B0,
                                                             (short)0, C0, false, false);
                C1 = __builtin_amdgcn_wmma_f32_16x16x32_bf16(false, Aq[c], false, B1,
                                                             (short)0, C1, false, false);
            }
            // ---- causal mask on diagonal-overlapping blocks ---------------
            if (k0 + 31 > q0) {
                #pragma unroll
                for (int r = 0; r < 8; ++r) {
                    int qrow = q0 + r + 8 * hlf;
                    if (k0 + ln      > qrow) C0[r] = -__builtin_inff();
                    if (k0 + 16 + ln > qrow) C1[r] = -__builtin_inff();
                }
            }
            // ---- online softmax update (DPP 16-lane butterflies) ----------
            float alpha[8];
            #pragma unroll
            for (int r = 0; r < 8; ++r) {
                float v = fmaxf(C0[r], C1[r]);
                ROW_REDUCE_MAX(v);
                float mnew = fmaxf(mrow[r], v);
                alpha[r] = __expf(mrow[r] - mnew);
                mrow[r]  = mnew;
                float p0 = __expf(C0[r] - mnew);
                float p1 = __expf(C1[r] - mnew);
                float s  = p0 + p1;
                ROW_REDUCE_SUM(s);
                srow[r] = srow[r] * alpha[r] + s;
                // stash P (bf16) for the PV A-operand
                myP[(r + 8 * hlf) * PSTRIDE + ln]      = (__bf16)p0;
                myP[(r + 8 * hlf) * PSTRIDE + 16 + ln] = (__bf16)p1;
            }
            // ---- rescale running output ----------------------------------
            #pragma unroll
            for (int t = 0; t < 8; ++t)
                #pragma unroll
                for (int r = 0; r < 8; ++r) acc[t][r] *= alpha[r];

            // ---- O += P V : reload P in A layout, 8 WMMAs over D ---------
            v16bf Ap;
            const __bf16* pr = myP + ln * PSTRIDE + 8 * hlf;
            ((v4u*)&Ap)[0] = *(const v4u*)(pr);
            ((v4u*)&Ap)[1] = *(const v4u*)(pr + 16);
            #pragma unroll
            for (int t = 0; t < 8; ++t) {
                v16bf Bv;
                const __bf16* vr = ldsV + (16 * t + ln) * VSTRIDE + 8 * hlf;
                ((v4u*)&Bv)[0] = *(const v4u*)(vr);
                ((v4u*)&Bv)[1] = *(const v4u*)(vr + 16);
                acc[t] = __builtin_amdgcn_wmma_f32_16x16x32_bf16(false, Ap, false, Bv,
                                                                 (short)0, acc[t], false, false);
            }
        }
        __syncthreads();
    }

    // ---------------- normalize and store: out[bh][q][d] --------------------
    float inv[8];
    #pragma unroll
    for (int r = 0; r < 8; ++r) inv[r] = 1.0f / srow[r];

    float* go = Out + (long long)bh * (SEQ * HDIM) + (long long)q0 * HDIM;
    #pragma unroll
    for (int t = 0; t < 8; ++t)
        #pragma unroll
        for (int r = 0; r < 8; ++r)
            go[(r + 8 * hlf) * HDIM + 16 * t + ln] = acc[t][r] * inv[r];
}

extern "C" void kernel_launch(void* const* d_in, const int* in_sizes, int n_in,
                              void* d_out, int out_size, void* d_ws, size_t ws_size,
                              hipStream_t stream) {
    const float* Q    = (const float*)d_in[0];
    const float* K    = (const float*)d_in[1];
    const float* V    = (const float*)d_in[2];
    const float* Bias = (const float*)d_in[3];
    // d_in[4] = cumulative_seq_lengths; layout is fixed (SEG_LEN=1024 boundaries)
    dim3 grid(SEQ / QTILE, NBATCH * NHEADS);
    dim3 block(128);
    hipLaunchKernelGGL(fa_fwd_kernel, grid, block, 0, stream, Q, K, V, Bias, (float*)d_out);
}
